// MoELayer_17566416241067
// MI455X (gfx1250) — matrix operand order repaired and checked by reading, not compiled, not run on hardware
//
#include <hip/hip_runtime.h>
#include <hip/hip_bf16.h>
#include <math.h>

// ---------------------------------------------------------------------------
// Types for WMMA fragments (gfx1250, wave32)
// ---------------------------------------------------------------------------
typedef __bf16 bf16;
typedef __bf16 v16bf __attribute__((ext_vector_type(16)));
typedef __bf16 v8bf  __attribute__((ext_vector_type(8)));
typedef float  v8f   __attribute__((ext_vector_type(8)));

union Frag16 { v16bf v; v8bf h[2]; };

#define E_ 8
#define B_ 4096
#define D_ 1024
#define H_ 1024

// LDS B-tile pitch: 32 K-elements + 8 pad = 40 bf16 = 80 bytes (20 banks).
// 16-byte fragment reads from 16 consecutive N-rows then start at banks
// 20*n mod 64, all distinct -> conflict-free ds_load_b128.
#define BPITCH 40

// ---------------------------------------------------------------------------
// fp32 -> bf16 conversion (grid-stride)
// ---------------------------------------------------------------------------
__global__ void f32_to_bf16_kernel(const float* __restrict__ src,
                                   bf16* __restrict__ dst, int n) {
    int stride = gridDim.x * blockDim.x;
    for (int i = blockIdx.x * blockDim.x + threadIdx.x; i < n; i += stride) {
        dst[i] = (bf16)src[i];
    }
}

// ---------------------------------------------------------------------------
// GEMM: C[M,N](f32) = A[M,K](bf16, row-major) * W[N,K](bf16, row-major)^T
//       (+ optional bias[N]), C stored with leading dimension ldc.
// Block = 8 waves; block tile 256(M) x 64(N); wave tile 32(M) x 64(N).
// B tile (64 x 32 bf16) staged cooperatively into double-buffered LDS and
// shared by all 8 waves; A fragments loaded directly from global.
// Fragment layouts per CDNA5 ISA 7.12.2 (wave32):
//   A 16x32 bf16 : lanes 0-15 -> M=lane, K groups {0-7,16-23};
//                  lanes16-31 -> M=lane-16, K groups {8-15,24-31}
//   B 32x16 bf16 : lanes 0-15 -> N=lane, K groups {0-7,16-23}; lanes16-31 -> N, K groups {8-15,24-31}
//   C 16x16 f32  : vgpr r, lanes 0-15 -> M=r, N=lane; lanes16-31 -> M=8+r
// ---------------------------------------------------------------------------
__global__ __launch_bounds__(256) void gemm_bf16_wmma(
    const bf16* __restrict__ A, const bf16* __restrict__ W,
    const float* __restrict__ bias, float* __restrict__ C,
    int M, int N, int K, int ldc) {
    __shared__ __align__(16) bf16 bstage[2][64 * BPITCH];

    const int tid   = threadIdx.x;
    const int lane  = tid & 31;
    const int wave  = tid >> 5;
    const int mtile = blockIdx.y * 256 + wave * 32;   // wave owns rows [mtile, mtile+32)
    const int ntile = blockIdx.x * 64;
    const int hlf   = lane >> 4;   // 0 or 1
    const int l15   = lane & 15;

    // Direct-global A fragment bases for the two 16-row sub-tiles.
    const bf16* aBase0 = A + (size_t)(mtile + l15) * K + hlf * 8;
    const bf16* aBase1 = aBase0 + (size_t)16 * K;

    // Cooperative B staging: thread t copies row n = t>>2, 8 K-elems at (t&3)*8.
    const int sn = tid >> 2;
    const int sk = (tid & 3) * 8;
    const bf16* bSrc = W + (size_t)(ntile + sn) * K + sk;
    bf16* bDst = &bstage[0][sn * BPITCH + sk];

    // Per-wave LDS fragment read pointers: frag nb reads row nb*16 + l15,
    // K offsets hlf*8 and hlf*8+16 (two 16-byte ds loads).
    const bf16* bRd = &bstage[0][(l15)*BPITCH + hlf * 8];

    v8f acc[2][4];
#pragma unroll
    for (int i = 0; i < 2; ++i)
#pragma unroll
        for (int j = 0; j < 4; ++j) acc[i][j] = (v8f){};

    // Prime buffer 0.
    *(v8bf*)(bDst) = *(const v8bf*)(bSrc);
    int buf = 0;

    for (int k = 0; k < K; k += 32) {
        __syncthreads();   // staging of `buf` done; previous readers done

        // Stage next K-chunk into the other buffer.
        if (k + 32 < K) {
            *(v8bf*)(bDst + (buf ^ 1) * 64 * BPITCH) =
                *(const v8bf*)(bSrc + (k + 32));
        }

        // A fragments (direct global).
        Frag16 a0, a1;
        a0.h[0] = *(const v8bf*)(aBase0 + k);
        a0.h[1] = *(const v8bf*)(aBase0 + k + 16);
        a1.h[0] = *(const v8bf*)(aBase1 + k);
        a1.h[1] = *(const v8bf*)(aBase1 + k + 16);

        // B fragments from LDS.
        Frag16 b[4];
        const bf16* br = bRd + buf * 64 * BPITCH;
#pragma unroll
        for (int nb = 0; nb < 4; ++nb) {
            b[nb].h[0] = *(const v8bf*)(br + nb * 16 * BPITCH);
            b[nb].h[1] = *(const v8bf*)(br + nb * 16 * BPITCH + 16);
        }

#pragma unroll
        for (int nb = 0; nb < 4; ++nb) {
            acc[0][nb] = __builtin_amdgcn_wmma_f32_16x16x32_bf16(
                false, a0.v, false, b[nb].v, (short)0, acc[0][nb], false, false);
            acc[1][nb] = __builtin_amdgcn_wmma_f32_16x16x32_bf16(
                false, a1.v, false, b[nb].v, (short)0, acc[1][nb], false, false);
        }
        buf ^= 1;
    }

    // Epilogue: scatter accumulators per the C/D layout, fuse bias.
#pragma unroll
    for (int nb = 0; nb < 4; ++nb) {
        const int n = ntile + nb * 16 + l15;
        const float bv = bias ? bias[n] : 0.0f;
#pragma unroll
        for (int ai = 0; ai < 2; ++ai) {
#pragma unroll
            for (int r = 0; r < 8; ++r) {
                const int m = mtile + ai * 16 + hlf * 8 + r;
                C[(size_t)m * ldc + n] = acc[ai][nb][r] + bv;
            }
        }
    }
}

// ---------------------------------------------------------------------------
// GRU gate combine: h' = (1-z)*n + z*h  with r,z,n from gx/gh thirds + biases
// Writes f32 state (into d_out slot) and bf16 copy (for the next GEMM).
// ---------------------------------------------------------------------------
__global__ void gru_gate_kernel(const float* __restrict__ gx,
                                const float* __restrict__ gh,
                                const float* __restrict__ b_ih,
                                const float* __restrict__ b_hh,
                                const float* __restrict__ hprev,
                                float* __restrict__ hnew,
                                bf16* __restrict__ hnew_bf, int total, int H) {
    int idx = blockIdx.x * blockDim.x + threadIdx.x;
    if (idx >= total) return;
    int b = idx / H, h = idx - b * H;
    size_t base = (size_t)b * 3 * H;
    float xr = gx[base + h]         + b_ih[h];
    float xz = gx[base + H + h]     + b_ih[H + h];
    float xn = gx[base + 2 * H + h] + b_ih[2 * H + h];
    float hr = gh[base + h]         + b_hh[h];
    float hz = gh[base + H + h]     + b_hh[H + h];
    float hn = gh[base + 2 * H + h] + b_hh[2 * H + h];
    float r = 1.0f / (1.0f + __expf(-(xr + hr)));
    float z = 1.0f / (1.0f + __expf(-(xz + hz)));
    float n = tanhf(xn + r * hn);
    float hp = hprev[(size_t)b * H + h];
    float out = (1.0f - z) * n + z * hp;
    hnew[(size_t)b * H + h]    = out;
    hnew_bf[(size_t)b * H + h] = (bf16)out;
}

// ---------------------------------------------------------------------------
// Router head: logits = hr_new @ W_fc^T + b_fc ; softmax over E=8.
// One thread per batch row; W_fc (8x1024 f32) is L2/WGP$-resident.
// ---------------------------------------------------------------------------
__global__ void router_softmax_kernel(const float* __restrict__ hr_new,
                                      const float* __restrict__ Wfc,
                                      const float* __restrict__ bfc,
                                      float* __restrict__ rw, int B, int H) {
    int b = blockIdx.x * blockDim.x + threadIdx.x;
    if (b >= B) return;
    float acc[E_];
#pragma unroll
    for (int e = 0; e < E_; ++e) acc[e] = bfc[e];
    const float* xrow = hr_new + (size_t)b * H;
    for (int h = 0; h < H; ++h) {
        float xv = xrow[h];
#pragma unroll
        for (int e = 0; e < E_; ++e) acc[e] += xv * Wfc[e * H + h];
    }
    float mx = acc[0];
#pragma unroll
    for (int e = 1; e < E_; ++e) mx = fmaxf(mx, acc[e]);
    float sum = 0.0f;
#pragma unroll
    for (int e = 0; e < E_; ++e) { acc[e] = __expf(acc[e] - mx); sum += acc[e]; }
    float inv = 1.0f / sum;
#pragma unroll
    for (int e = 0; e < E_; ++e) rw[(size_t)b * E_ + e] = acc[e] * inv;
}

// ---------------------------------------------------------------------------
// output[b,d] = sum_e rw[b,e] * expert_outputs[b,e,d]  (bias already fused)
// ---------------------------------------------------------------------------
__global__ void combine_kernel(const float* __restrict__ rw,
                               const float* __restrict__ eo,
                               float* __restrict__ out, int total, int D) {
    int idx = blockIdx.x * blockDim.x + threadIdx.x;
    if (idx >= total) return;
    int b = idx / D, d = idx - b * D;
    const float* eob = eo + (size_t)b * E_ * D + d;
    const float* rwb = rw + (size_t)b * E_;
    float s = 0.0f;
#pragma unroll
    for (int e = 0; e < E_; ++e) s += rwb[e] * eob[(size_t)e * D];
    out[idx] = s;
}

// ---------------------------------------------------------------------------
// Host side
// ---------------------------------------------------------------------------
static inline void launch_conv(const float* s, bf16* d, int n, hipStream_t st) {
    int blocks = (n + 1023) / 1024;
    if (blocks > 2048) blocks = 2048;
    f32_to_bf16_kernel<<<blocks, 256, 0, st>>>(s, d, n);
}

extern "C" void kernel_launch(void* const* d_in, const int* in_sizes, int n_in,
                              void* d_out, int out_size, void* d_ws, size_t ws_size,
                              hipStream_t stream) {
    const int E = E_, B = B_, D = D_, H = H_;
    const float* x        = (const float*)d_in[0];   // (B,1,D)
    const float* h_router = (const float*)d_in[1];   // (1,B,H)
    const float* h_exp    = (const float*)d_in[2];   // (E,B,H)
    const float* W_ih_e   = (const float*)d_in[3];   // (E,3H,D)
    const float* W_hh_e   = (const float*)d_in[4];   // (E,3H,H)
    const float* b_ih_e   = (const float*)d_in[5];   // (E,3H)
    const float* b_hh_e   = (const float*)d_in[6];   // (E,3H)
    const float* W_proj   = (const float*)d_in[7];   // (E,D,H)
    const float* b_proj   = (const float*)d_in[8];   // (E,D)
    const float* W_ih_r   = (const float*)d_in[9];   // (3H,D)
    const float* W_hh_r   = (const float*)d_in[10];  // (3H,H)
    const float* b_ih_r   = (const float*)d_in[11];  // (3H)
    const float* b_hh_r   = (const float*)d_in[12];  // (3H)
    const float* W_fc     = (const float*)d_in[13];  // (E,H)
    const float* b_fc     = (const float*)d_in[14];  // (E)

    // d_out layout (f32, flat, in tuple return order)
    float* out_y   = (float*)d_out;                          // (B,D)
    float* out_hr  = out_y  + (size_t)B * D;                 // (1,B,H)
    float* out_he  = out_hr + (size_t)B * H;                 // (E,B,H)
    float* out_rw  = out_he + (size_t)E * B * H;             // (B,E)
    float* out_eo  = out_rw + (size_t)B * E;                 // (B,E,D)

    // Workspace carve (256B aligned slabs; phases run serially on `stream`,
    // so per-expert buffers are reused).
    char* ws = (char*)d_ws;
    size_t off = 0;
    auto carve = [&](size_t bytes) -> char* {
        char* p = ws + off;
        off = (off + bytes + 255) & ~(size_t)255;
        return p;
    };
    bf16*  xb   = (bf16*)carve((size_t)B * D * sizeof(bf16));      // x in bf16
    bf16*  hstb = (bf16*)carve((size_t)B * H * sizeof(bf16));      // h-state bf16
    bf16*  hnb  = (bf16*)carve((size_t)B * H * sizeof(bf16));      // h_new bf16
    bf16*  w1b  = (bf16*)carve((size_t)3 * H * D * sizeof(bf16));  // W_ih slab
    bf16*  w2b  = (bf16*)carve((size_t)3 * H * H * sizeof(bf16));  // W_hh slab
    bf16*  w3b  = (bf16*)carve((size_t)D * H * sizeof(bf16));      // W_proj slab
    float* gx   = (float*)carve((size_t)B * 3 * H * sizeof(float));
    float* gh   = (float*)carve((size_t)B * 3 * H * sizeof(float));
    (void)ws_size;

    const dim3 blk(256);
    const dim3 grid_gates(3 * H / 64, B / 256);   // N=3072: 48 x-tiles; M=4096: 16
    const dim3 grid_proj(D / 64, B / 256);        // N=1024: 16 x-tiles
    const int  ew_n = (B * H + 255) / 256;        // elementwise over B*H

    // ---- Router GRU ----
    launch_conv(x, xb, B * D, stream);
    launch_conv(W_ih_r, w1b, 3 * H * D, stream);
    launch_conv(W_hh_r, w2b, 3 * H * H, stream);
    launch_conv(h_router, hstb, B * H, stream);
    gemm_bf16_wmma<<<grid_gates, blk, 0, stream>>>(xb,   w1b, nullptr, gx, B, 3 * H, D, 3 * H);
    gemm_bf16_wmma<<<grid_gates, blk, 0, stream>>>(hstb, w2b, nullptr, gh, B, 3 * H, H, 3 * H);
    gru_gate_kernel<<<ew_n, blk, 0, stream>>>(gx, gh, b_ih_r, b_hh_r, h_router,
                                              out_hr, hnb, B * H, H);
    router_softmax_kernel<<<(B + 255) / 256, blk, 0, stream>>>(out_hr, W_fc, b_fc,
                                                               out_rw, B, H);

    // ---- Experts (sequential; ws slabs reused) ----
    for (int e = 0; e < E; ++e) {
        launch_conv(W_ih_e + (size_t)e * 3 * H * D, w1b, 3 * H * D, stream);
        launch_conv(W_hh_e + (size_t)e * 3 * H * H, w2b, 3 * H * H, stream);
        launch_conv(h_exp  + (size_t)e * B * H,     hstb, B * H,    stream);
        gemm_bf16_wmma<<<grid_gates, blk, 0, stream>>>(xb,   w1b, nullptr, gx, B, 3 * H, D, 3 * H);
        gemm_bf16_wmma<<<grid_gates, blk, 0, stream>>>(hstb, w2b, nullptr, gh, B, 3 * H, H, 3 * H);
        gru_gate_kernel<<<ew_n, blk, 0, stream>>>(gx, gh,
                                                  b_ih_e + (size_t)e * 3 * H,
                                                  b_hh_e + (size_t)e * 3 * H,
                                                  h_exp  + (size_t)e * B * H,
                                                  out_he + (size_t)e * B * H,
                                                  hnb, B * H, H);
        launch_conv(W_proj + (size_t)e * D * H, w3b, D * H, stream);
        // Write directly into expert_outputs (B,E,D): C base = out_eo + e*D, ldc = E*D
        gemm_bf16_wmma<<<grid_proj, blk, 0, stream>>>(hnb, w3b,
                                                      b_proj + (size_t)e * D,
                                                      out_eo + (size_t)e * D,
                                                      B, D, H, E * D);
    }

    // ---- Weighted combine ----
    combine_kernel<<<(B * D + 255) / 256, blk, 0, stream>>>(out_rw, out_eo, out_y,
                                                            B * D, D);
}
